// QuantBasicTransformerBlockINT4_9706626089382
// MI455X (gfx1250) — matrix operand order repaired and checked by reading, not compiled, not run on hardware
//
#include <hip/hip_runtime.h>
#include <math.h>
#include <stdint.h>

typedef unsigned short u16;
typedef unsigned int   u32;
typedef unsigned long long u64;
typedef __attribute__((ext_vector_type(16))) __bf16 v16bf;
typedef __attribute__((ext_vector_type(8)))  float  v8f;
typedef __attribute__((ext_vector_type(4)))  u32    su4;
typedef __attribute__((ext_vector_type(8)))  u32    su8;

#define DIMD   1280
#define CTXD   768
#define FFI    5120
#define ROWS   8192      // 8*1024
#define CROWS  616       // 8*77
#define NHEAD  20
#define DH     64

// ---------------- common device helpers ----------------

__device__ __forceinline__ u16 f32_bf16(float f) {
  union { float f; u32 u; } a; a.f = f;
  u32 r = a.u + 0x7FFFu + ((a.u >> 16) & 1u);
  return (u16)(r >> 16);
}

struct Frag { u32 u[8]; };
union FragV { Frag f; v16bf v; };

__device__ __forceinline__ v8f wmma_bf16(const Frag& a, const Frag& b, v8f c) {
  FragV ua, ub; ua.f = a; ub.f = b;
  return __builtin_amdgcn_wmma_f32_16x16x32_bf16(false, ua.v, false, ub.v,
                                                 (short)0, c, false, false);
}

// A-matrix 16x32 bf16 K mapping: VGPR v -> K=2v (+16 for lane-half)
__device__ __forceinline__ int kmapA(int v) { return 2*v; }
// B-matrix: VGPR v -> K=2v (+16 for lane-half)
__device__ __forceinline__ int kmapB(int v) { return 2*v; }

// ---------------- Tensor Data Mover: 2D tile global -> LDS ----------------
// data_size = 2 bytes (bf16). Pads 4 DWORDs after every 16 DWORDs stored, so a
// 32-element (64B) tile row lands with an 80B LDS stride (== SA*2 below).
// tensor_dim1 clips rows: reads past it return zero (handles M-tail blocks).

__device__ __forceinline__ void tdm_issue_2d_bf16(
    u32 lds_off, const void* gaddr, u32 tensor_d0, u32 tensor_d1,
    u32 tile_d0, u32 tile_d1, u32 stride_elems, int pad)
{
  u64 ga = (u64)(uintptr_t)gaddr;
  su4 g0;
  g0[0] = 1u;                                           // count=1, gather off
  g0[1] = lds_off;                                      // LDS byte address
  g0[2] = (u32)ga;                                      // global_addr[31:0]
  g0[3] = ((u32)(ga >> 32) & 0x01FFFFFFu) | (2u << 30); // addr[56:32] | type=2
  u32 w0 = (1u << 16);                                  // data_size = 2 bytes
  if (pad) w0 |= (1u << 20) | (3u << 22) | (3u << 25);  // pad: every 16 dw, +4 dw
  su8 g1;
  g1[0] = w0;
  g1[1] = (tensor_d0 & 0xFFFFu) << 16;                  // tensor_dim0[15:0]
  g1[2] = (tensor_d0 >> 16) | ((tensor_d1 & 0xFFFFu) << 16);
  g1[3] = (tensor_d1 >> 16) | (tile_d0 << 16);          // tile_dim0
  g1[4] = tile_d1 & 0xFFFFu;                            // tile_dim1 (tile_dim2=0)
  g1[5] = stride_elems;                                 // tensor_dim0_stride[31:0]
  g1[6] = 0;                                            // stride hi / dim1_stride lo
  g1[7] = 0;
  asm volatile("tensor_load_to_lds %0, %1" :: "s"(g0), "s"(g1) : "memory");
}

// ---------------- fp32 -> bf16 convert ----------------

__global__ void cvt_bf16_kernel(const float* __restrict__ in, u16* __restrict__ out, int n) {
  int i = blockIdx.x * 256 + threadIdx.x;
  if (i < n) out[i] = f32_bf16(in[i]);
}

// ---------------- LayerNorm (fp32 in, bf16 out) ----------------

__global__ __launch_bounds__(256) void layernorm_bf16(
    const float* __restrict__ x, const float* __restrict__ g,
    const float* __restrict__ bta, u16* __restrict__ out, int D)
{
  __shared__ float red[256];
  int row = blockIdx.x, t = threadIdx.x;
  const float* xr = x + (size_t)row * D;
  float s = 0.f, ss = 0.f;
  for (int c = t; c < D; c += 256) { float v = xr[c]; s += v; ss += v * v; }
  red[t] = s; __syncthreads();
  for (int o = 128; o > 0; o >>= 1) { if (t < o) red[t] += red[t + o]; __syncthreads(); }
  float mean = red[0] * (1.0f / D); __syncthreads();
  red[t] = ss; __syncthreads();
  for (int o = 128; o > 0; o >>= 1) { if (t < o) red[t] += red[t + o]; __syncthreads(); }
  float var = red[0] * (1.0f / D) - mean * mean;
  float inv = rsqrtf(var + 1e-5f);
  for (int c = t; c < D; c += 256)
    out[(size_t)row * D + c] = f32_bf16((xr[c] - mean) * inv * g[c] + bta[c]);
}

// ---------------- WMMA GEMM (TDM-fed A tile) ----------------
// C[M,N] = A[M,K](bf16) * B[K,N](bf16)  [+bias] [+resid] ; out f32 or bf16.
// GEGLU=1: out_bf16 = (C_a + bias[n]) * gelu(C_g + bias[n+gOff]) with C_g from B cols n+gOff.

#define GM 128
#define GN 64
#define SA 40
#define SB 40

template<int GEGLU>
__global__ __launch_bounds__(256) void gemm_wmma(
    const u16* __restrict__ A, const u16* __restrict__ B,
    const float* __restrict__ bias, const float* resid,
    float* outF, u16* outH,
    int M, int N, int K, int ldb, int gOff)
{
  __shared__ u16 As[GM * SA];                 // A tile, [m][k], 80B row stride (TDM pad)
  __shared__ u16 Bs[GN * SB];                 // B tile transposed, [n][k]
  __shared__ u16 Bs2[GEGLU ? GN * SB : 16];

  int t = threadIdx.x;
  int bm = blockIdx.x * GM, bn = blockIdx.y * GN;
  int wave = t >> 5, lane = t & 31, lm = lane & 15, lh = lane >> 4;
  int mw = wave >> 1, nw = wave & 1;          // 4x2 wave grid, 32x32 per wave

  v8f acc[2][2] = {};
  v8f acc2[2][2] = {};

  int bkr = t >> 3, bn0 = (t & 7) * 8;        // B loader: k-row, n-part
  u32 asLds = (u32)(uintptr_t)&As[0];

  for (int k0 = 0; k0 < K; k0 += 32) {
    union { uint4 q; u32 u[4]; u16 s[8]; } bq, bq2;
    bq.q = *(const uint4*)(B + (size_t)(k0 + bkr) * ldb + bn + bn0);
    if (GEGLU) bq2.q = *(const uint4*)(B + (size_t)(k0 + bkr) * ldb + bn + gOff + bn0);

    __syncthreads();  // previous iteration's LDS reads done
    if (wave == 0) {
      // DMA the 128x32 A tile (row stride K) into As with 16B/row LDS padding.
      tdm_issue_2d_bf16(asLds, A + (size_t)bm * K + k0,
                        (u32)K, (u32)(M - bm), 32u, (u32)GM, (u32)K, 1);
    }
    #pragma unroll
    for (int i = 0; i < 8; ++i) Bs[(bn0 + i) * SB + bkr] = bq.s[i];
    if (GEGLU) {
      #pragma unroll
      for (int i = 0; i < 8; ++i) Bs2[(bn0 + i) * SB + bkr] = bq2.s[i];
    }
    if (wave == 0) __builtin_amdgcn_s_wait_tensorcnt(0);
    __syncthreads();

    Frag fa[2], fb[2], fb2[2];
    #pragma unroll
    for (int f = 0; f < 2; ++f)
      #pragma unroll
      for (int v = 0; v < 8; ++v)
        fa[f].u[v] = *(const u32*)&As[(mw * 32 + f * 16 + lm) * SA + kmapA(v) + 16 * lh];
    #pragma unroll
    for (int f = 0; f < 2; ++f)
      #pragma unroll
      for (int v = 0; v < 8; ++v)
        fb[f].u[v] = *(const u32*)&Bs[(nw * 32 + f * 16 + lm) * SB + kmapB(v) + 16 * lh];
    if (GEGLU) {
      #pragma unroll
      for (int f = 0; f < 2; ++f)
        #pragma unroll
        for (int v = 0; v < 8; ++v)
          fb2[f].u[v] = *(const u32*)&Bs2[(nw * 32 + f * 16 + lm) * SB + kmapB(v) + 16 * lh];
    }
    #pragma unroll
    for (int i = 0; i < 2; ++i)
      #pragma unroll
      for (int j = 0; j < 2; ++j) {
        acc[i][j] = wmma_bf16(fa[i], fb[j], acc[i][j]);
        if (GEGLU) acc2[i][j] = wmma_bf16(fa[i], fb2[j], acc2[i][j]);
      }
  }

  #pragma unroll
  for (int i = 0; i < 2; ++i)
    #pragma unroll
    for (int j = 0; j < 2; ++j)
      #pragma unroll
      for (int r = 0; r < 8; ++r) {
        int gr = bm + mw * 32 + i * 16 + r + 8 * lh;
        int gc = bn + nw * 32 + j * 16 + lm;
        if (gr < M) {
          float c = acc[i][j][r];
          if (GEGLU) {
            float av = c + bias[gc];
            float gv = acc2[i][j][r] + bias[gc + gOff];
            float gel = 0.5f * gv * (1.0f + erff(gv * 0.70710678118654752f));
            outH[(size_t)gr * N + gc] = f32_bf16(av * gel);
          } else {
            if (bias)  c += bias[gc];
            if (resid) c += resid[(size_t)gr * N + gc];
            if (outF)  outF[(size_t)gr * N + gc] = c;
            else       outH[(size_t)gr * N + gc] = f32_bf16(c);
          }
        }
      }
}

// ---------------- two-pass flash attention (bf16 in/out, WMMA) ----------------
// One block: (batch b, head h, 32 query rows). Key chunks of 256, masked at Jvalid.

__global__ __launch_bounds__(256) void attn_wmma(
    const u16* __restrict__ Q, const u16* __restrict__ Kp,
    const u16* __restrict__ Vp, u16* __restrict__ O,
    int Jvalid, int kvBS)
{
  __shared__ u16   Qs[32 * 72];
  __shared__ float Sc[32 * 257];
  __shared__ u16   Pc[32 * 264];
  __shared__ float sm[32], sl[32], slinv[32];

  const float scale = 0.125f;  // dh^-0.5, dh=64
  int t = threadIdx.x, wave = t >> 5, lane = t & 31;
  int lm = lane & 15, lh = lane >> 4;
  int qt = blockIdx.x & 31;
  int h  = (blockIdx.x >> 5) % NHEAD;
  int b  = blockIdx.x / (32 * NHEAD);

  size_t qbase  = ((size_t)b * 1024 + (size_t)qt * 32) * DIMD + h * DH;
  size_t kvbase = (size_t)b * kvBS * DIMD + h * DH;

  { int r = t >> 3, c0 = (t & 7) * 8;
    *(uint4*)&Qs[r * 72 + c0] = *(const uint4*)(Q + qbase + (size_t)r * DIMD + c0); }
  if (t < 32) { sm[t] = -1e30f; sl[t] = 0.f; }
  __syncthreads();

  int mt = wave & 1, nset = wave >> 1;
  Frag qa[2];
  #pragma unroll
  for (int ks = 0; ks < 2; ++ks)
    #pragma unroll
    for (int v = 0; v < 8; ++v)
      qa[ks].u[v] = *(const u32*)&Qs[(mt * 16 + lm) * 72 + ks * 32 + kmapA(v) + 16 * lh];

  int nChunks = (Jvalid + 255) >> 8;

  // ---- pass A: row max / sum(exp) over all keys ----
  for (int ch = 0; ch < nChunks; ++ch) {
    int j0 = ch << 8;
    #pragma unroll
    for (int nn = 0; nn < 4; ++nn) {
      int nt = nset * 4 + nn;
      v8f s = {};
      #pragma unroll
      for (int ks = 0; ks < 2; ++ks) {
        Frag kb;
        #pragma unroll
        for (int v = 0; v < 8; ++v) {
          int j  = j0 + nt * 16 + lm;
          int kk = ks * 32 + kmapB(v) + 16 * lh;
          kb.u[v] = *(const u32*)(Kp + kvbase + (size_t)j * DIMD + kk);
        }
        s = wmma_bf16(qa[ks], kb, s);
      }
      #pragma unroll
      for (int r = 0; r < 8; ++r)
        Sc[(mt * 16 + r + 8 * lh) * 257 + nt * 16 + lm] = s[r];
    }
    __syncthreads();
    if (t < 32) {
      float mo = sm[t], lo = sl[t], cm = -1e30f;
      for (int c = 0; c < 256; ++c)
        if (j0 + c < Jvalid) cm = fmaxf(cm, Sc[t * 257 + c] * scale);
      float mn = fmaxf(mo, cm);
      float l  = lo * __expf(mo - mn);
      for (int c = 0; c < 256; ++c)
        if (j0 + c < Jvalid) l += __expf(Sc[t * 257 + c] * scale - mn);
      sm[t] = mn; sl[t] = l;
    }
    __syncthreads();
  }
  if (t < 32) slinv[t] = 1.0f / sl[t];
  __syncthreads();

  // ---- pass B: P = softmax tiles (bf16), O += P @ V ----
  int dt = wave >> 1;   // dh tile 0..3
  v8f oacc = {};
  for (int ch = 0; ch < nChunks; ++ch) {
    int j0 = ch << 8;
    #pragma unroll
    for (int nn = 0; nn < 4; ++nn) {
      int nt = nset * 4 + nn;
      v8f s = {};
      #pragma unroll
      for (int ks = 0; ks < 2; ++ks) {
        Frag kb;
        #pragma unroll
        for (int v = 0; v < 8; ++v) {
          int j  = j0 + nt * 16 + lm;
          int kk = ks * 32 + kmapB(v) + 16 * lh;
          kb.u[v] = *(const u32*)(Kp + kvbase + (size_t)j * DIMD + kk);
        }
        s = wmma_bf16(qa[ks], kb, s);
      }
      #pragma unroll
      for (int r = 0; r < 8; ++r) {
        int row = mt * 16 + r + 8 * lh;
        int col = j0 + nt * 16 + lm;
        float p = (col < Jvalid) ? __expf(s[r] * scale - sm[row]) * slinv[row] : 0.f;
        Pc[row * 264 + nt * 16 + lm] = f32_bf16(p);
      }
    }
    __syncthreads();
    #pragma unroll
    for (int ks = 0; ks < 8; ++ks) {
      Frag pa, vb;
      #pragma unroll
      for (int v = 0; v < 8; ++v)
        pa.u[v] = *(const u32*)&Pc[(mt * 16 + lm) * 264 + ks * 32 + kmapA(v) + 16 * lh];
      #pragma unroll
      for (int v = 0; v < 8; ++v) {
        int j = j0 + ks * 32 + kmapB(v) + 16 * lh;
        size_t base = kvbase + (size_t)j * DIMD + dt * 16 + lm;
        u32 lo32 = Vp[base], hi32 = Vp[base + DIMD];
        vb.u[v] = lo32 | (hi32 << 16);
      }
      oacc = wmma_bf16(pa, vb, oacc);
    }
    __syncthreads();
  }

  #pragma unroll
  for (int r = 0; r < 8; ++r) {
    size_t grow = (size_t)b * 1024 + qt * 32 + mt * 16 + r + 8 * lh;
    int gcol = h * DH + dt * 16 + lm;
    O[grow * DIMD + gcol] = f32_bf16(oacc[r]);
  }
}

// ---------------- host orchestration ----------------

extern "C" void kernel_launch(void* const* d_in, const int* in_sizes, int n_in,
                              void* d_out, int out_size, void* d_ws, size_t ws_size,
                              hipStream_t stream) {
  (void)in_sizes; (void)n_in; (void)out_size; (void)ws_size;
  const float* x    = (const float*)d_in[0];
  const float* ctx  = (const float*)d_in[1];
  const float* n1g  = (const float*)d_in[3];
  const float* n1b  = (const float*)d_in[4];
  const float* q1   = (const float*)d_in[5];
  const float* k1   = (const float*)d_in[6];
  const float* v1   = (const float*)d_in[7];
  const float* o1w  = (const float*)d_in[8];
  const float* o1bi = (const float*)d_in[9];
  const float* n2g  = (const float*)d_in[10];
  const float* n2b  = (const float*)d_in[11];
  const float* q2   = (const float*)d_in[12];
  const float* k2   = (const float*)d_in[13];
  const float* v2   = (const float*)d_in[14];
  const float* o2w  = (const float*)d_in[15];
  const float* o2bi = (const float*)d_in[16];
  const float* n3g  = (const float*)d_in[17];
  const float* n3b  = (const float*)d_in[18];
  const float* w1   = (const float*)d_in[19];
  const float* b1   = (const float*)d_in[20];
  const float* w2   = (const float*)d_in[21];
  const float* b2   = (const float*)d_in[22];

  char* W = (char*)d_ws; size_t off = 0;
  auto alloc = [&](size_t bytes) -> char* {
    off = (off + 255) & ~(size_t)255;
    char* p = W + off; off += bytes; return p;
  };
  u16* q1b  = (u16*)alloc((size_t)DIMD * DIMD * 2);
  u16* k1b  = (u16*)alloc((size_t)DIMD * DIMD * 2);
  u16* v1b  = (u16*)alloc((size_t)DIMD * DIMD * 2);
  u16* o1b  = (u16*)alloc((size_t)DIMD * DIMD * 2);
  u16* q2b  = (u16*)alloc((size_t)DIMD * DIMD * 2);
  u16* o2b  = (u16*)alloc((size_t)DIMD * DIMD * 2);
  u16* k2b  = (u16*)alloc((size_t)CTXD * DIMD * 2);
  u16* v2b  = (u16*)alloc((size_t)CTXD * DIMD * 2);
  u16* w1b  = (u16*)alloc((size_t)DIMD * 2 * FFI * 2);
  u16* w2b  = (u16*)alloc((size_t)FFI * DIMD * 2);
  u16* ctxb = (u16*)alloc((size_t)CROWS * CTXD * 2);
  u16* hb   = (u16*)alloc((size_t)ROWS * DIMD * 2);
  u16* Qb   = (u16*)alloc((size_t)ROWS * DIMD * 2);
  u16* Kb   = (u16*)alloc((size_t)ROWS * DIMD * 2);
  u16* Vb   = (u16*)alloc((size_t)ROWS * DIMD * 2);
  u16* atb  = (u16*)alloc((size_t)ROWS * DIMD * 2);
  u16* ffb  = (u16*)alloc((size_t)ROWS * FFI * 2);
  float* xcur = (float*)alloc((size_t)ROWS * DIMD * 4);

  auto cvt = [&](const float* src, u16* dst, size_t n) {
    cvt_bf16_kernel<<<dim3((unsigned)((n + 255) / 256)), 256, 0, stream>>>(src, dst, (int)n);
  };
  cvt(q1, q1b, (size_t)DIMD * DIMD);   cvt(k1, k1b, (size_t)DIMD * DIMD);
  cvt(v1, v1b, (size_t)DIMD * DIMD);   cvt(o1w, o1b, (size_t)DIMD * DIMD);
  cvt(q2, q2b, (size_t)DIMD * DIMD);   cvt(o2w, o2b, (size_t)DIMD * DIMD);
  cvt(k2, k2b, (size_t)CTXD * DIMD);   cvt(v2, v2b, (size_t)CTXD * DIMD);
  cvt(w1, w1b, (size_t)DIMD * 2 * FFI); cvt(w2, w2b, (size_t)FFI * DIMD);
  cvt(ctx, ctxb, (size_t)CROWS * CTXD);

  hipMemcpyAsync(xcur, x, (size_t)ROWS * DIMD * 4, hipMemcpyDeviceToDevice, stream);

  auto gemm = [&](const u16* A, const u16* B, const float* bias, const float* resid,
                  float* oF, u16* oH, int M, int N, int K, int ldb) {
    dim3 g((M + GM - 1) / GM, N / GN);
    gemm_wmma<0><<<g, 256, 0, stream>>>(A, B, bias, resid, oF, oH, M, N, K, ldb, 0);
  };

  // ---- self-attention ----
  layernorm_bf16<<<ROWS, 256, 0, stream>>>(xcur, n1g, n1b, hb, DIMD);
  gemm(hb, q1b, nullptr, nullptr, nullptr, Qb, ROWS, DIMD, DIMD, DIMD);
  gemm(hb, k1b, nullptr, nullptr, nullptr, Kb, ROWS, DIMD, DIMD, DIMD);
  gemm(hb, v1b, nullptr, nullptr, nullptr, Vb, ROWS, DIMD, DIMD, DIMD);
  attn_wmma<<<8 * NHEAD * 32, 256, 0, stream>>>(Qb, Kb, Vb, atb, 1024, 1024);
  gemm(atb, o1b, o1bi, xcur, xcur, nullptr, ROWS, DIMD, DIMD, DIMD);

  // ---- cross-attention ----
  layernorm_bf16<<<ROWS, 256, 0, stream>>>(xcur, n2g, n2b, hb, DIMD);
  gemm(hb, q2b, nullptr, nullptr, nullptr, Qb, ROWS, DIMD, DIMD, DIMD);
  gemm(ctxb, k2b, nullptr, nullptr, nullptr, Kb, CROWS, DIMD, CTXD, DIMD);
  gemm(ctxb, v2b, nullptr, nullptr, nullptr, Vb, CROWS, DIMD, CTXD, DIMD);
  attn_wmma<<<8 * NHEAD * 32, 256, 0, stream>>>(Qb, Kb, Vb, atb, 77, 77);
  gemm(atb, o2b, o2bi, xcur, xcur, nullptr, ROWS, DIMD, DIMD, DIMD);

  // ---- GEGLU feedforward ----
  layernorm_bf16<<<ROWS, 256, 0, stream>>>(xcur, n3g, n3b, hb, DIMD);
  {
    dim3 g((ROWS + GM - 1) / GM, FFI / GN);
    gemm_wmma<1><<<g, 256, 0, stream>>>(hb, w1b, b1, nullptr, nullptr, ffb,
                                        ROWS, FFI, DIMD, 2 * FFI, FFI);
  }
  gemm(ffb, w2b, b2, xcur, (float*)d_out, nullptr, ROWS, DIMD, FFI, DIMD);
}
